// TrainableLDAHead_54116587929768
// MI455X (gfx1250) — matrix operand order repaired and checked by reading, not compiled, not run on hardware
//
#include <hip/hip_runtime.h>

// ---------------------------------------------------------------------------
// Problem constants (from the reference)
// ---------------------------------------------------------------------------
#define N_PTS   16384
#define N_CLS   10000
#define N_DIM   128
#define N_CLSP  10112   // 79 * 128, zero-padded class count for clean tiling

typedef __attribute__((ext_vector_type(16))) __bf16 v16bf;
typedef __attribute__((ext_vector_type(8)))  __bf16 v8bf;
typedef __attribute__((ext_vector_type(8)))  float  v8f;

// ---------------------------------------------------------------------------
// bf16 helpers (round-to-nearest-even split: x ~= hi + lo, lo ~ 2^-16 rel err)
// ---------------------------------------------------------------------------
static __device__ __forceinline__ unsigned short f32_to_bf16_rne(float f) {
    unsigned int u = __float_as_uint(f);
    unsigned int r = u + 0x7FFFu + ((u >> 16) & 1u);
    return (unsigned short)(r >> 16);
}
static __device__ __forceinline__ float bf16_to_f32(unsigned short h) {
    return __uint_as_float(((unsigned int)h) << 16);
}

// ---------------------------------------------------------------------------
// Pack rows of a [rows x 128] fp32 matrix into bf16 hi/lo + row squared norms.
// One block (128 threads) per row; rows >= srcRows are zero-filled padding.
// ---------------------------------------------------------------------------
__global__ __launch_bounds__(128)
void pack_rows(const float* __restrict__ src, int srcRows,
               unsigned short* __restrict__ hi, unsigned short* __restrict__ lo,
               float* __restrict__ sq) {
    __shared__ float red[128];
    const int row = blockIdx.x;
    const int d   = threadIdx.x;
    float x = 0.0f;
    if (row < srcRows) x = src[(size_t)row * N_DIM + d];
    const unsigned short h = f32_to_bf16_rne(x);
    const unsigned short l = f32_to_bf16_rne(x - bf16_to_f32(h));
    const size_t idx = (size_t)row * N_DIM + d;
    hi[idx] = h;
    lo[idx] = l;
    red[d] = x * x;
    __syncthreads();
#pragma unroll
    for (int s = 64; s > 0; s >>= 1) {
        if (d < s) red[d] += red[d + s];
        __syncthreads();
    }
    if (d == 0) sq[row] = red[0];
}

// ---------------------------------------------------------------------------
// logsumexp over prior logits (single block)
// ---------------------------------------------------------------------------
__global__ __launch_bounds__(256)
void lse_kernel(const float* __restrict__ logits, int n, float* __restrict__ out) {
    __shared__ float red[256];
    const int t = threadIdx.x;
    float mx = -3.402823466e+38f;
    for (int i = t; i < n; i += 256) mx = fmaxf(mx, logits[i]);
    red[t] = mx;
    __syncthreads();
#pragma unroll
    for (int s = 128; s > 0; s >>= 1) {
        if (t < s) red[t] = fmaxf(red[t], red[t + s]);
        __syncthreads();
    }
    const float gmax = red[0];
    __syncthreads();
    float sum = 0.0f;
    for (int i = t; i < n; i += 256) sum += __expf(logits[i] - gmax);
    red[t] = sum;
    __syncthreads();
#pragma unroll
    for (int s = 128; s > 0; s >>= 1) {
        if (t < s) red[t] += red[t + s];
        __syncthreads();
    }
    if (t == 0) out[0] = gmax + __logf(red[0]);
}

// ---------------------------------------------------------------------------
// Main fused GEMM + LDA epilogue.
//   grid  = (N_PTS/128, N_CLSP/128), block = 256 (8 waves of 32)
//   wave w handles rows [blk*128 + 16w, +16), all 128 columns of the block
//   cross accumulated with 3x bf16 WMMA (hi*hi + hi*lo + lo*hi)
// ---------------------------------------------------------------------------
__global__ __launch_bounds__(256)
void lda_gemm_wmma(const unsigned short* __restrict__ zh,
                   const unsigned short* __restrict__ zl,
                   const unsigned short* __restrict__ mh,
                   const unsigned short* __restrict__ ml,
                   const float* __restrict__ zsq,
                   const float* __restrict__ musq,
                   const float* __restrict__ prior,
                   const float* __restrict__ lse,
                   const float* __restrict__ log_cov,
                   float* __restrict__ out) {
    const int lane = threadIdx.x & 31;
    const int wave = threadIdx.x >> 5;
    const int nm   = lane & 15;   // A-row index within tile == B-col index
    const int kh   = lane >> 4;   // K-half selector (CDNA5 wave32 WMMA layout)

    const int mBase = blockIdx.x * 128 + wave * 16;
    const int cBase = blockIdx.y * 128;

    v8f acc[8] = {};

    const unsigned short* zrh = zh + (size_t)(mBase + nm) * N_DIM;
    const unsigned short* zrl = zl + (size_t)(mBase + nm) * N_DIM;

#pragma unroll
    for (int k = 0; k < N_DIM; k += 32) {
        // A fragment (16x32 bf16): lane holds K = [k+16*kh, +16) contiguous
        const int ka0 = k + 16 * kh;
        const int ka1 = k + 16 * kh + 8;
        const v8bf ah0 = *(const v8bf*)(zrh + ka0);
        const v8bf ah1 = *(const v8bf*)(zrh + ka1);
        const v8bf al0 = *(const v8bf*)(zrl + ka0);
        const v8bf al1 = *(const v8bf*)(zrl + ka1);
        const v16bf Ah = __builtin_shufflevector(ah0, ah1,
            0,1,2,3,4,5,6,7,8,9,10,11,12,13,14,15);
        const v16bf Al = __builtin_shufflevector(al0, al1,
            0,1,2,3,4,5,6,7,8,9,10,11,12,13,14,15);

#pragma unroll
        for (int t = 0; t < 8; ++t) {
            // B fragment (32x16 bf16): lane = col, holds K = [k+16*kh, +16) contiguous
            const size_t boff = (size_t)(cBase + t * 16 + nm) * N_DIM + (k + 16 * kh);
            const v16bf Bh = *(const v16bf*)(mh + boff);
            const v16bf Bl = *(const v16bf*)(ml + boff);
            acc[t] = __builtin_amdgcn_wmma_f32_16x16x32_bf16(
                false, Ah, false, Bh, (short)0, acc[t], false, false);
            acc[t] = __builtin_amdgcn_wmma_f32_16x16x32_bf16(
                false, Ah, false, Bl, (short)0, acc[t], false, false);
            acc[t] = __builtin_amdgcn_wmma_f32_16x16x32_bf16(
                false, Al, false, Bh, (short)0, acc[t], false, false);
        }
    }

    // ---- epilogue: out = log_prior - 0.5*(m2/var + D*lc) -------------------
    const float lc        = log_cov[0];
    const float inv_var   = __expf(-lc);
    const float half_ldet = 0.5f * (float)N_DIM * lc;
    const float ls        = lse[0];

    // C/D layout: VGPR r, lanes 0-15 -> M=r, lanes 16-31 -> M=r+8; N = lane&15
    const int rbase = mBase + (kh << 3);
    float zs[8];
#pragma unroll
    for (int r = 0; r < 8; ++r) zs[r] = zsq[rbase + r];

#pragma unroll
    for (int t = 0; t < 8; ++t) {
        const int col = cBase + t * 16 + nm;
        if (col >= N_CLS) continue;   // only last column-block partially valid
        const float msq = musq[col];
        const float lp  = prior[col] - ls;
#pragma unroll
        for (int r = 0; r < 8; ++r) {
            const float m2 = zs[r] - 2.0f * acc[t][r] + msq;
            out[(size_t)(rbase + r) * N_CLS + col] =
                lp - 0.5f * m2 * inv_var - half_ldet;
        }
    }
}

// ---------------------------------------------------------------------------
// Host launcher
// ---------------------------------------------------------------------------
extern "C" void kernel_launch(void* const* d_in, const int* in_sizes, int n_in,
                              void* d_out, int out_size, void* d_ws, size_t ws_size,
                              hipStream_t stream) {
    (void)in_sizes; (void)n_in; (void)out_size; (void)ws_size;
    const float* z       = (const float*)d_in[0];   // [N, D]
    const float* mu      = (const float*)d_in[1];   // [C, D]
    const float* log_cov = (const float*)d_in[2];   // [1]
    const float* prior   = (const float*)d_in[3];   // [C]
    float*       out     = (float*)d_out;           // [N, C]

    // Workspace carve-up (~13.7 MB total)
    unsigned short* zh  = (unsigned short*)d_ws;
    unsigned short* zl  = zh + (size_t)N_PTS * N_DIM;
    unsigned short* mh  = zl + (size_t)N_PTS * N_DIM;
    unsigned short* ml  = mh + (size_t)N_CLSP * N_DIM;
    float*          zsq = (float*)(ml + (size_t)N_CLSP * N_DIM);
    float*          msq = zsq + N_PTS;
    float*          lse = msq + N_CLSP;

    pack_rows<<<dim3(N_PTS),  dim3(128), 0, stream>>>(z,  N_PTS, zh, zl, zsq);
    pack_rows<<<dim3(N_CLSP), dim3(128), 0, stream>>>(mu, N_CLS, mh, ml, msq);
    lse_kernel<<<dim3(1), dim3(256), 0, stream>>>(prior, N_CLS, lse);

    lda_gemm_wmma<<<dim3(N_PTS / 128, N_CLSP / 128), dim3(256), 0, stream>>>(
        zh, zl, mh, ml, zsq, msq, prior, lse, log_cov, out);
}
